// AttentionPool2d_3049426780703
// MI455X (gfx1250) — compile-verified, matrix operand order checked by reading, not run on hardware
//
#include <hip/hip_runtime.h>
#include <hip/hip_bf16.h>
#include <math.h>

// AttentionPool2d on MI455X (gfx1250, wave32, WMMA + Tensor Data Mover).
//
// Only out[:, :, 0] is returned -> only the t=0 query row is needed ->
// single-query flash attention per (batch, head). Dominant work is the K/V
// projection GEMM (M=1024, K=512, N=32*1025 ~ 34 GFLOP) fused with online
// softmax so k/v never touch HBM. bf16 WMMA (f32 accumulate) puts compute at
// ~10-15us; mandatory HBM traffic is x (64MiB) + packed bf16 xf (~72MiB r/w)
// ~= 6us at 23.3 TB/s -> near the matrix-op roofline.
//
// CDNA5-specific data path: xf and w_kv are pre-packed in HBM as bf16 in
// *fragment order* (K-groups permuted [0..7,16..23,8..15,24..31]); the flash
// kernel stages every 128x32 tile with TENSOR_LOAD_TO_LDS (2-D D#: 64B rows,
// 128 rows, LDS pad 16B per 64B -> 80B stride), double-buffered on TENSORcnt.
// Each WMMA fragment is exactly 2x ds_load_b128; all 8 B-fragments are
// preloaded so the 8 WMMAs issue back-to-back behind a single dscnt wait.

typedef float  v8f   __attribute__((ext_vector_type(8)));
typedef __bf16 v16bf __attribute__((ext_vector_type(16)));
typedef __bf16 v8bf  __attribute__((ext_vector_type(8)));
typedef unsigned int v4u __attribute__((ext_vector_type(4)));
typedef int v4i __attribute__((ext_vector_type(4)));
typedef int v8i __attribute__((ext_vector_type(8)));

#define EMBED   512
#define HEADS   8
#define CH      64
#define TTOK    1025
#define TPAD    1152          // 9 tiles * 128 token rows (zero padded)
#define NTILE   128
#define NTILES  9
#define NKB     16            // 512 / 32 K-chunks
#define NSTAGE  (NTILES * NKB)
#define LROW    40            // LDS row stride in bf16 (64B data + 16B pad)

// K permutation within a 32-block so a lane's 16 fragment values are
// contiguous: stored position p <- source offset so(p); groups 1 and 2 swap.
__device__ __forceinline__ int kperm_src(int p) {
    const int g = p >> 3;
    return (g == 1 || g == 2) ? (p ^ 24) : p;
}

// ---------------------------------------------------------------------------
// TDM issue: 2-D tile, rows of 64B (data_size=8B, tile_dim0=8 units),
// tile_dim1=128 rows, row stride `stride8` (in 8B units), LDS padding
// 16B after every 64B (pad_interval=3 -> 64B, pad_amount=3 -> 16B).
// ---------------------------------------------------------------------------
__device__ __forceinline__ v8i tdm_g1(int stride8) {
    v8i g = {};
    g[0] = (3 << 16) | (1 << 20) | (3 << 22) | (3 << 25); // dsize=8B,pad,int=64B,amt=16B
    g[1] = (int)(8u << 16);     // tensor_dim0 = 8 units (bits 79:48 low half)
    g[2] = (int)(128u << 16);   // tensor_dim1 = 128     (bits 111:80 low half)
    g[3] = (int)(8u << 16);     // tile_dim0 = 8         (bits 127:112)
    g[4] = 128;                 // tile_dim1 = 128       (bits 143:128)
    g[5] = stride8;             // tensor_dim0_stride    (bits 207:160 low 32)
    return g;
}

__device__ __forceinline__ void tdm_load(unsigned lds_addr, const void* gptr, v8i g1) {
    const unsigned long long ga = (unsigned long long)(uintptr_t)gptr;
    v4u g0;
    g0[0] = 1u;                                   // count=1, user descriptor
    g0[1] = lds_addr;                             // bits 63:32
    g0[2] = (unsigned)ga;                         // global_addr low
    g0[3] = (unsigned)((ga >> 32) & 0x01FFFFFFu)  // global_addr[56:32]
            | (2u << 30);                         // type = 2 ("image")
    const v4i z4 = {};
#if __clang_major__ >= 23
    const v8i z8 = {};
    __builtin_amdgcn_tensor_load_to_lds(g0, g1, z4, z4, z8, 0);
#else
    __builtin_amdgcn_tensor_load_to_lds(g0, g1, z4, z4, 0);
#endif
}

// ---------------------------------------------------------------------------
// Kernel 1: spatial mean of x (reads the full 64 MiB once, coalesced).
// ---------------------------------------------------------------------------
__global__ __launch_bounds__(256)
void mean_kernel(const float* __restrict__ x, float* __restrict__ mean)
{
    const int b = blockIdx.x, grp = blockIdx.y;
    const int lane = threadIdx.x & 31, wave = threadIdx.x >> 5;
#pragma unroll
    for (int i = 0; i < 4; ++i) {
        const int c = grp * 32 + wave * 4 + i;
        const float* p = x + (((size_t)b * EMBED + c) << 10);
        float s = 0.f;
#pragma unroll
        for (int j = 0; j < 32; ++j) s += p[lane + (j << 5)];
#pragma unroll
        for (int off = 16; off > 0; off >>= 1) s += __shfl_xor(s, off, 32);
        if (lane == 0) mean[b * EMBED + c] = s * (1.0f / 1024.0f);
    }
}

// ---------------------------------------------------------------------------
// Kernel 2: pack w_kv -> bf16, per-(head, K-block) fragment-ready 8KB tiles:
// wbf[(h*16+kb)*128 + row][32 bf16 permuted]. grid (8, 16), block 256.
// ---------------------------------------------------------------------------
__global__ __launch_bounds__(256)
void pack_w_kernel(const float* __restrict__ w_qkv, __bf16* __restrict__ wbf)
{
    const int h = blockIdx.x, kb = blockIdx.y, tid = threadIdx.x;
    const int row = tid >> 1, half = tid & 1;
    const int ch = (row < CH) ? (EMBED + h * CH + row)
                              : (2 * EMBED + h * CH + (row - CH));
    const float* src = w_qkv + (size_t)ch * EMBED + kb * 32;
    __bf16* dst = wbf + (((size_t)(h * NKB + kb)) * 128 + row) * 32 + half * 16;
#pragma unroll
    for (int i = 0; i < 16; ++i) {
        const int p = half * 16 + i;
        dst[i] = (__bf16)src[kperm_src(p)];
    }
}

// ---------------------------------------------------------------------------
// Kernel 3: pack xf -> bf16, token-major, K-permuted, zero-padded to 1152
// rows: xfT[b][t][512]. LDS-transposed so reads (along t) and writes
// (64B rows) are both contiguous. grid (32, 9), block 256.
// ---------------------------------------------------------------------------
__global__ __launch_bounds__(256)
void pack_xf_kernel(const float* __restrict__ x, const float* __restrict__ pos_emb,
                    const float* __restrict__ mean, __bf16* __restrict__ xfT)
{
    const int b = blockIdx.x;
    const int t0 = blockIdx.y * NTILE;
    const int tid = threadIdx.x;
    __shared__ float tile[64][129];

    for (int cb = 0; cb < 8; ++cb) {
        __syncthreads();
        {   // load 64c x 128t, coalesced along t
            const int cl = tid >> 2;
            const int tb = (tid & 3) << 5;
            const int cg = cb * 64 + cl;
            const float* xp = x + (((size_t)b * EMBED + cg) << 10);
            const float* pe = pos_emb + (size_t)cg * TTOK;
            const float  mv = mean[b * EMBED + cg];
            for (int i = 0; i < 32; ++i) {
                const int tl = tb + i;
                const int t  = t0 + tl;
                float v;
                if      (t == 0)   v = mv + pe[0];
                else if (t < TTOK) v = xp[t - 1] + pe[t];
                else               v = 0.f;
                tile[cl][tl] = v;
            }
        }
        __syncthreads();
        {   // store transposed + permuted: 64B contiguous per thread
            const int tl = tid >> 1, half = tid & 1;
            __bf16* dst = xfT + ((size_t)b * TPAD + (t0 + tl)) * EMBED
                              + cb * 64 + half * 32;
#pragma unroll
            for (int p = 0; p < 32; ++p)
                dst[p] = (__bf16)tile[half * 32 + kperm_src(p)][tl];
        }
    }
}

// ---------------------------------------------------------------------------
// Kernel 4: q0[b,c] = b_q[c] + w_q[c,:].(mean[b,:] + pos_emb[:,0]). 17 MFLOP.
// ---------------------------------------------------------------------------
__global__ __launch_bounds__(256)
void q0_kernel(const float* __restrict__ pos_emb, const float* __restrict__ w_qkv,
               const float* __restrict__ b_qkv,  const float* __restrict__ mean,
               float* __restrict__ q0)
{
    const int b = blockIdx.x, tid = threadIdx.x;
    __shared__ float xf0[EMBED];
    for (int k = tid; k < EMBED; k += 256)
        xf0[k] = mean[b * EMBED + k] + pos_emb[(size_t)k * TTOK];
    __syncthreads();
    for (int c = tid; c < EMBED; c += 256) {
        float s = b_qkv[c];
        const float* wp = w_qkv + (size_t)c * EMBED;
#pragma unroll 8
        for (int k = 0; k < EMBED; ++k) s += wp[k] * xf0[k];
        q0[b * EMBED + c] = s;
    }
}

// ---------------------------------------------------------------------------
// Kernel 5: fused K/V projection + single-query flash attention.
// One block per (head, batch); 8 wave32s. TDM double-buffers the 128x32 A/B
// tiles into 80B-stride LDS; per K-chunk each wave preloads A + 8 B fragments
// (2x ds_load_b128 each), then issues 8 v_wmma_f32_16x16x32_bf16 back-to-back.
// ---------------------------------------------------------------------------
__global__ __launch_bounds__(256)
void attn_flash_kernel(const __bf16* __restrict__ wbf, const __bf16* __restrict__ xfT,
                       const float* __restrict__ b_qkv, const float* __restrict__ q0,
                       float* __restrict__ a0)
{
    const int h    = blockIdx.x;
    const int b    = blockIdx.y;
    const int tid  = threadIdx.x;
    const int lane = tid & 31;
    const int wave = tid >> 5;
    const int rlo  = lane & 15;
    const int hi16 = lane & 16;          // fragment half select (bf16 elements)

    __shared__ __align__(16) __bf16 Abuf[2][128 * LROW];
    __shared__ __align__(16) __bf16 Bbuf[2][128 * LROW];
    __shared__ float kvs[128][133];
    __shared__ float q0s[CH], bias_s[128], accum_s[CH];
    __shared__ float logit_s[NTILE], p_s[NTILE];
    __shared__ float m_s, l_s, alpha_s, newm_s;

    if (tid < CH) { q0s[tid] = q0[b * EMBED + h * CH + tid]; accum_s[tid] = 0.f; }
    if (tid < 128) {
        const int ch = (tid < CH) ? (EMBED + h * CH + tid)
                                  : (2 * EMBED + h * CH + (tid - CH));
        bias_s[tid] = b_qkv[ch];
    }
    if (tid == 0) { m_s = -INFINITY; l_s = 0.f; }
    __syncthreads();

    const unsigned aLds[2] = { (unsigned)(uintptr_t)&Abuf[0][0],
                               (unsigned)(uintptr_t)&Abuf[1][0] };
    const unsigned bLds[2] = { (unsigned)(uintptr_t)&Bbuf[0][0],
                               (unsigned)(uintptr_t)&Bbuf[1][0] };
    const v8i g1A = tdm_g1(8);     // A rows contiguous (64B apart)
    const v8i g1B = tdm_g1(128);   // B rows 1024B apart

    // prologue: stage 0 into buffer 0
    if (wave == 0) {
        tdm_load(aLds[0], wbf + (size_t)(h * NKB) * 4096, g1A);
        tdm_load(bLds[0], xfT + (size_t)b * TPAD * EMBED, g1B);
    }

    int stage = 0;
    for (int tile = 0; tile < NTILES; ++tile) {
        const int t0 = tile * NTILE;
        v8f acc[8] = {};

        for (int kci = 0; kci < NKB; ++kci, ++stage) {
            const int cur = stage & 1;
            const int nxt = stage + 1;
            if (wave == 0) {
                if (nxt < NSTAGE) {
                    const int nt = nxt >> 4, nk = nxt & 15;
                    tdm_load(aLds[cur ^ 1], wbf + (size_t)(h * NKB + nk) * 4096, g1A);
                    tdm_load(bLds[cur ^ 1],
                             xfT + ((size_t)b * TPAD + nt * NTILE) * EMBED + nk * 32, g1B);
                    __builtin_amdgcn_s_wait_tensorcnt(2); // current stage landed
                } else {
                    __builtin_amdgcn_s_wait_tensorcnt(0);
                }
            }
            __syncthreads();

            const __bf16* Ab = &Abuf[cur][0];
            const __bf16* Bb = &Bbuf[cur][0];
            // preload A fragment + all 8 B fragments, then 8 WMMAs back-to-back
            const v8bf* ap = (const v8bf*)(Ab + (wave * 16 + rlo) * LROW + hi16);
            const v16bf af = __builtin_shufflevector(ap[0], ap[1],
                0,1,2,3,4,5,6,7,8,9,10,11,12,13,14,15);
            v16bf bfr[8];
#pragma unroll
            for (int n = 0; n < 8; ++n) {
                const v8bf* bp = (const v8bf*)(Bb + (n * 16 + rlo) * LROW + hi16);
                bfr[n] = __builtin_shufflevector(bp[0], bp[1],
                    0,1,2,3,4,5,6,7,8,9,10,11,12,13,14,15);
            }
#pragma unroll
            for (int n = 0; n < 8; ++n)
                acc[n] = __builtin_amdgcn_wmma_f32_16x16x32_bf16(
                    false, af, false, bfr[n], (short)0, acc[n], false, false);
            __syncthreads();
        }

        // spill k/v tile (+bias). D layout: M = r + 8*(lane>=16), N = lane&15
        {
            const int hi8 = hi16 >> 1;
#pragma unroll
            for (int n = 0; n < 8; ++n)
#pragma unroll
                for (int r = 0; r < 8; ++r) {
                    const int row = wave * 16 + r + hi8;
                    kvs[row][n * 16 + rlo] = acc[n][r] + bias_s[row];
                }
        }
        __syncthreads();

        // logits: (q0 . k_t) / 8 ; rows 0..63 of kvs are k-channels
        if (tid < NTILE) {
            const int t = t0 + tid;
            float lg = -INFINITY;
            if (t < TTOK) {
                float d = 0.f;
#pragma unroll 8
                for (int c = 0; c < CH; ++c) d += q0s[c] * kvs[c][tid];
                lg = d * 0.125f;      // scale^2 = 1/sqrt(64)
            }
            logit_s[tid] = lg;
        }
        __syncthreads();

        if (wave == 0) {
            float mx = fmaxf(fmaxf(logit_s[lane],      logit_s[lane + 32]),
                             fmaxf(logit_s[lane + 64], logit_s[lane + 96]));
#pragma unroll
            for (int off = 16; off > 0; off >>= 1) mx = fmaxf(mx, __shfl_xor(mx, off, 32));
            if (lane == 0) {
                const float nm = fmaxf(m_s, mx);
                newm_s  = nm;
                alpha_s = (m_s == -INFINITY) ? 0.f : __expf(m_s - nm);
                m_s     = nm;
            }
        }
        __syncthreads();
        if (tid < NTILE) {
            const float lg = logit_s[tid];
            p_s[tid] = (lg == -INFINITY) ? 0.f : __expf(lg - newm_s);
        }
        __syncthreads();
        if (wave == 0) {
            float sm = p_s[lane] + p_s[lane + 32] + p_s[lane + 64] + p_s[lane + 96];
#pragma unroll
            for (int off = 16; off > 0; off >>= 1) sm += __shfl_xor(sm, off, 32);
            if (lane == 0) l_s = alpha_s * l_s + sm;
        }
        if (tid < CH) {   // acc[c] = alpha*acc[c] + sum_t p_t * v[c,t]
            float a = alpha_s * accum_s[tid];
#pragma unroll 8
            for (int t = 0; t < NTILE; ++t) a += p_s[t] * kvs[CH + tid][t];
            accum_s[tid] = a;
        }
        __syncthreads();
    }

    if (tid < CH) a0[(size_t)b * EMBED + h * CH + tid] = accum_s[tid] / l_s;
}

// ---------------------------------------------------------------------------
// Kernel 6: out[b,o] = b_c[o] + w_c[o,:].a0[b,:]. 17 MFLOP.
// ---------------------------------------------------------------------------
__global__ __launch_bounds__(256)
void proj_kernel(const float* __restrict__ w_c, const float* __restrict__ b_c,
                 const float* __restrict__ a0,  float* __restrict__ out)
{
    const int b = blockIdx.x, tid = threadIdx.x;
    __shared__ float as[EMBED];
    for (int c = tid; c < EMBED; c += 256) as[c] = a0[b * EMBED + c];
    __syncthreads();
    for (int o = tid; o < EMBED; o += 256) {
        float s = b_c[o];
        const float* wp = w_c + (size_t)o * EMBED;
#pragma unroll 8
        for (int c = 0; c < EMBED; ++c) s += wp[c] * as[c];
        out[b * EMBED + o] = s;
    }
}

extern "C" void kernel_launch(void* const* d_in, const int* in_sizes, int n_in,
                              void* d_out, int out_size, void* d_ws, size_t ws_size,
                              hipStream_t stream)
{
    const float* x       = (const float*)d_in[0];  // [32, 512, 32, 32]
    const float* pos_emb = (const float*)d_in[1];  // [512, 1025]
    const float* w_qkv   = (const float*)d_in[2];  // [1536, 512]
    const float* b_qkv   = (const float*)d_in[3];  // [1536]
    const float* w_c     = (const float*)d_in[4];  // [512, 512]
    const float* b_c     = (const float*)d_in[5];  // [512]
    float* out = (float*)d_out;                    // [32, 512]

    // workspace: 3x 64KB f32 + 1MB packed weights + 36MB packed xf (~38MB)
    float*  wsf  = (float*)d_ws;
    float*  mean = wsf;                    // 32*512
    float*  q0   = wsf + 32 * EMBED;       // 32*512
    float*  a0   = wsf + 64 * EMBED;       // 32*512
    __bf16* wbf  = (__bf16*)(wsf + 96 * EMBED);              // 8*16*128*32
    __bf16* xfT  = (__bf16*)((char*)d_ws + 96 * EMBED * 4 + 1048576);

    mean_kernel<<<dim3(32, 16), 256, 0, stream>>>(x, mean);
    pack_w_kernel<<<dim3(HEADS, NKB), 256, 0, stream>>>(w_qkv, wbf);
    pack_xf_kernel<<<dim3(32, NTILES), 256, 0, stream>>>(x, pos_emb, mean, xfT);
    q0_kernel<<<32, 256, 0, stream>>>(pos_emb, w_qkv, b_qkv, mean, q0);
    attn_flash_kernel<<<dim3(HEADS, 32), 256, 0, stream>>>(wbf, xfT, b_qkv, q0, a0);
    proj_kernel<<<32, 256, 0, stream>>>(w_c, b_c, a0, out);
}
